// Attention_59047210385633
// MI455X (gfx1250) — matrix operand order repaired and checked by reading, not compiled, not run on hardware
//
#include <hip/hip_runtime.h>
#include <hip/hip_bf16.h>

// ---------------- problem constants ----------------
constexpr int B_   = 2;
constexpr int N_   = 2305;          // 48*48 + 1
constexpr int D_   = 768;
constexpr int H_   = 12;
constexpr int E_   = 64;
constexpr int N1_  = 48;
constexpr int N2_  = 48;
constexpr int NPAD = 2336;          // 73 * 32  (multiple of 32 for key blocks)
constexpr int MT   = NPAD / 16;     // 146 row tiles
constexpr int KST  = D_ / 32;       // 24 k-steps of 32 for D=768 contractions
constexpr int NB   = NPAD / 32;     // 73 key blocks

typedef unsigned short u16;
typedef __attribute__((ext_vector_type(16))) __bf16 v16bf;
typedef __attribute__((ext_vector_type(8)))  float  v8f;

union FragU { v16bf v; uint4 q[2]; };
union Pack8 { u16 u[8]; uint4 q; };

// ---------------- helpers ----------------
__device__ inline u16 f2bf(float f) {
    union { float f; unsigned u; } x; x.f = f;
    unsigned r = x.u + 0x7FFFu + ((x.u >> 16) & 1u);   // round-to-nearest-even
    return (u16)(r >> 16);
}

__device__ inline v8f zero8() {
    v8f z = {0.f,0.f,0.f,0.f,0.f,0.f,0.f,0.f};
    return z;
}

// A fragment (16x32 bf16, MxK): lane holds row m = lane&15; half = lane>>4.
// elems 0..15 <-> K = kofs + 16*half + 0..15 contiguous
__device__ inline v16bf ldA(const u16* p, int ldk, int kofs, int lane) {
    int m = lane & 15, half = lane >> 4;
    const u16* row = p + (size_t)m * ldk + kofs;
    FragU f;
    f.q[0] = *reinterpret_cast<const uint4*>(row + 16 * half);
    f.q[1] = *reinterpret_cast<const uint4*>(row + 16 * half + 8);
    return f.v;
}

// B fragment (32x16 bf16, KxN) from [N][K] row-major storage:
// lane holds column n = lane&15; K = kofs + 16*(lane>>4) + 0..15 contiguous.
__device__ inline v16bf ldB(const u16* p, int ldk, int kofs, int lane) {
    int n = lane & 15, kb2 = (lane >> 4) * 16;
    const u16* row = p + (size_t)n * ldk + kofs + kb2;
    FragU f;
    f.q[0] = *reinterpret_cast<const uint4*>(row);
    f.q[1] = *reinterpret_cast<const uint4*>(row + 8);
    return f.v;
}

__device__ inline v8f wmma_bf16(v16bf a, v16bf b, v8f c) {
    return __builtin_amdgcn_wmma_f32_16x16x32_bf16(false, a, false, b,
                                                   (short)0, c, false, false);
}

__device__ inline float hmax16(float v) {
    #pragma unroll
    for (int m = 1; m < 16; m <<= 1) v = fmaxf(v, __shfl_xor(v, m, 32));
    return v;
}
__device__ inline float hsum16(float v) {
    #pragma unroll
    for (int m = 1; m < 16; m <<= 1) v += __shfl_xor(v, m, 32);
    return v;
}

// ---------------- kernel 1: fp32 -> bf16 converts ----------------
__global__ void k_convert_x(const float* __restrict__ x, u16* __restrict__ xb) {
    int idx = blockIdx.x * blockDim.x + threadIdx.x;
    int total = B_ * NPAD * D_;
    if (idx >= total) return;
    int c = idx % D_;
    int t = idx / D_;
    int n = t % NPAD;
    int b = t / NPAD;
    float v = (n < N_) ? x[((size_t)b * N_ + n) * D_ + c] : 0.f;
    xb[idx] = f2bf(v);
}

__global__ void k_convert_w(const float* __restrict__ w, u16* __restrict__ wb) {
    int idx = blockIdx.x * blockDim.x + threadIdx.x;
    if (idx >= D_ * D_) return;
    wb[idx] = f2bf(w[idx]);
}

// ---------------- kernel 2: fused QKV projection + 2D RoPE ----------------
// grid: (h=12, mt=146, b=2), block: 32 (one wave -> 16 rows x one head).
// 12 accumulator tiles (q/k/v x 4 col-tiles) stay live (96 VGPRs); each
// A-fragment is loaded once per k-step and feeds 12 independent WMMAs.
__global__ void k_qkv(const u16* __restrict__ xb,
                      const u16* __restrict__ wqb, const u16* __restrict__ wkb,
                      const u16* __restrict__ wvb,
                      const float* __restrict__ pos0, const float* __restrict__ pos1,
                      u16* __restrict__ qb, u16* __restrict__ kbuf,
                      u16* __restrict__ vtb) {
    int lane = threadIdx.x;
    int h  = blockIdx.x;
    int mt = blockIdx.y;
    int b  = blockIdx.z;
    int n = lane & 15, half = lane >> 4;

    const u16* arow = xb + ((size_t)b * NPAD + mt * 16) * D_;
    const u16* wbase0 = wqb + (size_t)h * E_ * D_;
    const u16* wbase1 = wkb + (size_t)h * E_ * D_;
    const u16* wbase2 = wvb + (size_t)h * E_ * D_;

    v8f acc[12];
    #pragma unroll
    for (int i = 0; i < 12; ++i) acc[i] = zero8();

    for (int ks = 0; ks < KST; ++ks) {
        int ko = ks * 32;
        v16bf af = ldA(arow, D_, ko, lane);
        #pragma unroll
        for (int nt = 0; nt < 4; ++nt) {
            acc[0 + nt] = wmma_bf16(af, ldB(wbase0 + (size_t)nt * 16 * D_, D_, ko, lane), acc[0 + nt]);
            acc[4 + nt] = wmma_bf16(af, ldB(wbase1 + (size_t)nt * 16 * D_, D_, ko, lane), acc[4 + nt]);
            acc[8 + nt] = wmma_bf16(af, ldB(wbase2 + (size_t)nt * 16 * D_, D_, ko, lane), acc[8 + nt]);
        }
    }

    // ---- epilogue: RoPE on q/k, transpose-store v ----
    #pragma unroll
    for (int mat = 0; mat < 2; ++mat) {
        u16* dst = (mat == 0) ? qb : kbuf;
        #pragma unroll
        for (int nt = 0; nt < 4; ++nt) {
            v8f a = acc[mat * 4 + nt];
            int e = nt * 16 + n;
            #pragma unroll
            for (int r = 0; r < 8; ++r) {
                int mrow = r + 8 * half;
                int ntok = mt * 16 + mrow;
                float val = a[r];
                float partner = __shfl_xor(val, 1, 32);   // pair (2j,2j+1) exchange
                float outv = val;
                if (ntok >= 1 && ntok < N_) {
                    int idx = ntok - 1;
                    int i1 = idx / N2_, i2 = idx % N2_;
                    int tj = (e & 31) >> 1;               // theta index
                    float p = (e < 32) ? pos0[b * N1_ + i1] : pos1[b * N2_ + i2];
                    float th = __powf(10000.f, -(float)tj * (1.0f / 16.0f));
                    float s, c;
                    __sincosf(p * th, &s, &c);
                    outv = ((e & 1) == 0) ? (val * c - partner * s)
                                          : (val * c + partner * s);
                }
                if (mat == 0) outv *= 0.125f;             // 1/sqrt(E)
                dst[(((size_t)(b * H_ + h)) * NPAD + ntok) * E_ + e] = f2bf(outv);
            }
        }
    }
    #pragma unroll
    for (int nt = 0; nt < 4; ++nt) {
        v8f a = acc[8 + nt];
        int e = nt * 16 + n;
        Pack8 pk;
        #pragma unroll
        for (int r = 0; r < 8; ++r) pk.u[r] = f2bf(a[r]);
        size_t base = (((size_t)(b * H_ + h)) * E_ + e) * NPAD + mt * 16 + 8 * half;
        *reinterpret_cast<uint4*>(vtb + base) = pk.q;     // 8 consecutive keys/lane
    }
}

// ---------------- kernel 3: flash attention (one wave = 16 queries) ----------
// grid: (qt=146, bh=24), block: 32
__global__ void k_attn(const u16* __restrict__ qb, const u16* __restrict__ kbuf,
                       const u16* __restrict__ vtb, u16* __restrict__ aob) {
    __shared__ u16 Pt[16 * 32];     // probs tile bounce buffer (C->A transpose)
    int lane = threadIdx.x;
    int qt = blockIdx.x;
    int bh = blockIdx.y;
    int b = bh / H_, h = bh % H_;
    int n = lane & 15, half = lane >> 4;

    const u16* Q = qb   + (size_t)bh * NPAD * E_ + (size_t)qt * 16 * E_;
    const u16* K = kbuf + (size_t)bh * NPAD * E_;
    const u16* V = vtb  + (size_t)bh * E_ * NPAD;

    v16bf qf0 = ldA(Q, E_, 0,  lane);
    v16bf qf1 = ldA(Q, E_, 32, lane);

    v8f o0 = zero8(), o1 = zero8(), o2 = zero8(), o3 = zero8();
    float rm[8], rl[8];
    #pragma unroll
    for (int r = 0; r < 8; ++r) { rm[r] = -__builtin_inff(); rl[r] = 0.f; }

    for (int blk = 0; blk < NB; ++blk) {
        int key0 = blk * 32;
        const u16* kr0 = K + (size_t)key0 * E_;
        const u16* kr1 = K + (size_t)(key0 + 16) * E_;

        v8f s0 = zero8(), s1 = zero8();
        s0 = wmma_bf16(qf0, ldB(kr0, E_, 0,  lane), s0);
        s0 = wmma_bf16(qf1, ldB(kr0, E_, 32, lane), s0);
        s1 = wmma_bf16(qf0, ldB(kr1, E_, 0,  lane), s1);
        s1 = wmma_bf16(qf1, ldB(kr1, E_, 32, lane), s1);

        bool ok0 = (key0 + n) < N_;
        bool ok1 = (key0 + 16 + n) < N_;

        #pragma unroll
        for (int r = 0; r < 8; ++r) {
            float a0 = ok0 ? s0[r] : -__builtin_inff();
            float a1 = ok1 ? s1[r] : -__builtin_inff();
            float bm = hmax16(fmaxf(a0, a1));
            float nm = fmaxf(rm[r], bm);
            float sc = __expf(rm[r] - nm);
            float p0 = ok0 ? __expf(s0[r] - nm) : 0.f;
            float p1 = ok1 ? __expf(s1[r] - nm) : 0.f;
            float bs = hsum16(p0 + p1);
            rl[r] = rl[r] * sc + bs;
            rm[r] = nm;
            o0[r] *= sc; o1[r] *= sc; o2[r] *= sc; o3[r] *= sc;
            int mrow = r + 8 * half;
            Pt[mrow * 32 + n]      = f2bf(p0);
            Pt[mrow * 32 + 16 + n] = f2bf(p1);
        }
        __syncthreads();
        v16bf pf = ldA(Pt, 32, 0, lane);     // probs as A-fragment (16x32)
        __syncthreads();

        o0 = wmma_bf16(pf, ldB(V + (size_t)0  * NPAD, NPAD, key0, lane), o0);
        o1 = wmma_bf16(pf, ldB(V + (size_t)16 * NPAD, NPAD, key0, lane), o1);
        o2 = wmma_bf16(pf, ldB(V + (size_t)32 * NPAD, NPAD, key0, lane), o2);
        o3 = wmma_bf16(pf, ldB(V + (size_t)48 * NPAD, NPAD, key0, lane), o3);
    }

    // normalize + store [b, ntok, h*64 + e] as bf16 for the out-projection
    #pragma unroll
    for (int r = 0; r < 8; ++r) {
        float inv = 1.0f / rl[r];
        o0[r] *= inv; o1[r] *= inv; o2[r] *= inv; o3[r] *= inv;
    }
    #pragma unroll
    for (int r = 0; r < 8; ++r) {
        int mrow = r + 8 * half;
        size_t row = (size_t)b * NPAD + qt * 16 + mrow;
        size_t base = row * D_ + h * E_;
        aob[base + 0  + n] = f2bf(o0[r]);
        aob[base + 16 + n] = f2bf(o1[r]);
        aob[base + 32 + n] = f2bf(o2[r]);
        aob[base + 48 + n] = f2bf(o3[r]);
    }
}

// ---------------- kernel 4: output projection ----------------
// grid: (ct=12 col groups of 64, mt=146, b=2), block: 32.
// 4 live accumulators; A-fragment loaded once per k-step, feeds 4 WMMAs.
__global__ void k_oproj(const u16* __restrict__ aob, const u16* __restrict__ wob,
                        float* __restrict__ out) {
    int lane = threadIdx.x;
    int ct = blockIdx.x;
    int mt = blockIdx.y;
    int b  = blockIdx.z;
    int n = lane & 15, half = lane >> 4;

    const u16* arow = aob + ((size_t)b * NPAD + mt * 16) * D_;
    const u16* wbase = wob + (size_t)ct * 64 * D_;

    v8f acc[4];
    #pragma unroll
    for (int i = 0; i < 4; ++i) acc[i] = zero8();

    for (int ks = 0; ks < KST; ++ks) {
        int ko = ks * 32;
        v16bf af = ldA(arow, D_, ko, lane);
        #pragma unroll
        for (int nt = 0; nt < 4; ++nt)
            acc[nt] = wmma_bf16(af, ldB(wbase + (size_t)nt * 16 * D_, D_, ko, lane), acc[nt]);
    }

    #pragma unroll
    for (int nt = 0; nt < 4; ++nt) {
        #pragma unroll
        for (int r = 0; r < 8; ++r) {
            int ntok = mt * 16 + r + 8 * half;
            if (ntok < N_)
                out[((size_t)b * N_ + ntok) * D_ + ct * 64 + nt * 16 + n] = acc[nt][r];
        }
    }
}

// ---------------- launcher ----------------
extern "C" void kernel_launch(void* const* d_in, const int* in_sizes, int n_in,
                              void* d_out, int out_size, void* d_ws, size_t ws_size,
                              hipStream_t stream) {
    (void)in_sizes; (void)n_in; (void)out_size; (void)ws_size;
    const float* x    = (const float*)d_in[0];
    const float* pos0 = (const float*)d_in[1];
    const float* pos1 = (const float*)d_in[2];
    const float* wq   = (const float*)d_in[3];
    const float* wk   = (const float*)d_in[4];
    const float* wv   = (const float*)d_in[5];
    const float* wo   = (const float*)d_in[6];
    float* out = (float*)d_out;

    const size_t XB_B  = (size_t)B_ * NPAD * D_ * sizeof(u16);       // 7,176,192
    const size_t W_B   = (size_t)D_ * D_ * sizeof(u16);              // 1,179,648
    const size_t QKV_B = (size_t)B_ * H_ * NPAD * E_ * sizeof(u16);  // 7,176,192

    char* ws = (char*)d_ws;
    u16* xb  = (u16*)(ws);
    u16* wqb = (u16*)(ws + XB_B);
    u16* wkb = (u16*)(ws + XB_B + W_B);
    u16* wvb = (u16*)(ws + XB_B + 2 * W_B);
    u16* wob = (u16*)(ws + XB_B + 3 * W_B);
    u16* qb  = (u16*)(ws + XB_B + 4 * W_B);
    u16* kb  = (u16*)(ws + XB_B + 4 * W_B + QKV_B);
    u16* vtb = (u16*)(ws + XB_B + 4 * W_B + 2 * QKV_B);
    u16* aob = (u16*)(ws + XB_B + 4 * W_B + 3 * QKV_B);

    {
        int total = B_ * NPAD * D_;
        k_convert_x<<<(total + 255) / 256, 256, 0, stream>>>(x, xb);
        int wtot = D_ * D_;
        int g = (wtot + 255) / 256;
        k_convert_w<<<g, 256, 0, stream>>>(wq, wqb);
        k_convert_w<<<g, 256, 0, stream>>>(wk, wkb);
        k_convert_w<<<g, 256, 0, stream>>>(wv, wvb);
        k_convert_w<<<g, 256, 0, stream>>>(wo, wob);
    }

    k_qkv<<<dim3(H_, MT, B_), 32, 0, stream>>>(xb, wqb, wkb, wvb,
                                               pos0, pos1, qb, kb, vtb);
    k_attn<<<dim3(MT, B_ * H_), 32, 0, stream>>>(qb, kb, vtb, aob);
    k_oproj<<<dim3(D_ / 64, MT, B_), 32, 0, stream>>>(aob, wob, out);
}
